// AttentionPooling_16544214024629
// MI455X (gfx1250) — compile-verified, matrix operand order checked by reading, not run on hardware
//
#include <hip/hip_runtime.h>

#define H 256
#define LDST 264  // padded bf16 row stride for the LDS x-tile (264*2B = 528B, odd dword multiple of 4 -> conflict-free b128 reads)

typedef __attribute__((ext_vector_type(16))) __bf16 bf16x16;
typedef __attribute__((ext_vector_type(8)))  __bf16 bf16x8;
typedef __attribute__((ext_vector_type(8)))  float  f32x8;

__device__ __forceinline__ unsigned int pack_bf16x2(float a, float b) {
  unsigned int ua = __float_as_uint(a), ub = __float_as_uint(b);
  ua += 0x7FFFu + ((ua >> 16) & 1u);          // RNE
  ub += 0x7FFFu + ((ub >> 16) & 1u);
  return ((ua >> 16) & 0xFFFFu) | (ub & 0xFFFF0000u);
}

__device__ __forceinline__ float tanh_fast(float v) {
  // tanh(v) = 1 - 2/(exp(2v)+1); exact limits at +/-inf, ~1e-6 rel error
  float e = __expf(2.0f * v);
  return 1.0f - 2.0f / (e + 1.0f);
}

__device__ __forceinline__ bf16x16 ldfrag(const unsigned short* p) {
  // elements 0..7 = K kb..kb+7 ; elements 8..15 = K kb+16..kb+23 (ISA 16-bit A/B layout)
  bf16x8 lo = *(const bf16x8*)(p);
  bf16x8 hi = *(const bf16x8*)(p + 16);
  return __builtin_shufflevector(lo, hi, 0,1,2,3,4,5,6,7,8,9,10,11,12,13,14,15);
}

// ---------------- kernel 1: W1 [K][N] fp32 -> W1^T [N][K] bf16 ----------------
__global__ void convert_w1_kernel(const float* __restrict__ w1,
                                  unsigned short* __restrict__ w1t) {
  int k = blockIdx.x, n = threadIdx.x;
  unsigned int u = __float_as_uint(w1[k * H + n]);
  u += 0x7FFFu + ((u >> 16) & 1u);
  w1t[n * H + k] = (unsigned short)(u >> 16);
}

// ---------------- kernel 2: logits[i] = tanh(x@W1 + b1) @ W2 + b2 ----------------
__global__ __launch_bounds__(256, 2) void attn_logits_kernel(
    const float* __restrict__ x, const unsigned short* __restrict__ w1t,
    const float* __restrict__ b1, const float* __restrict__ w2,
    const float* __restrict__ b2, float* __restrict__ logits, int n_nodes) {
  __shared__ __align__(16) unsigned short xt[32 * LDST];
  __shared__ float lpart[8][32];

  const int tid = threadIdx.x;
  const int m0  = blockIdx.x * 32;

  // stage 32x256 fp32 rows of x into LDS as bf16 (coalesced float2 loads)
  #pragma unroll
  for (int it = 0; it < 16; ++it) {
    int p   = it * 256 + tid;          // 0..4095
    int row = p >> 7;                  // 0..31
    int cp  = p & 127;                 // dword-pair column
    int gr  = m0 + row;
    float a = 0.f, b = 0.f;
    if (gr < n_nodes) {
      const float2 v = ((const float2*)(x + (size_t)gr * H))[cp];
      a = v.x; b = v.y;
    }
    ((unsigned int*)xt)[row * (LDST / 2) + cp] = pack_bf16x2(a, b);
  }
  __syncthreads();

  const int wid   = tid >> 5;
  const int lane  = tid & 31;
  const int lrow  = lane & 15;     // matrix row/col within fragment
  const int khalf = lane >> 4;     // which K half this lane holds
  const int n0    = wid * 32;      // this wave's 32 output columns

  f32x8 acc[2][2];
  #pragma unroll
  for (int mb = 0; mb < 2; ++mb)
    #pragma unroll
    for (int nb = 0; nb < 2; ++nb)
      acc[mb][nb] = (f32x8){0.f,0.f,0.f,0.f,0.f,0.f,0.f,0.f};

  const unsigned short* abase0 = &xt[(lrow)      * LDST];
  const unsigned short* abase1 = &xt[(16 + lrow) * LDST];
  const unsigned short* wbase0 = w1t + (size_t)(n0 + lrow)      * H;
  const unsigned short* wbase1 = w1t + (size_t)(n0 + 16 + lrow) * H;

  #pragma unroll
  for (int k0 = 0; k0 < H; k0 += 32) {
    const int kb = k0 + khalf * 8;
    bf16x16 a0  = ldfrag(abase0 + kb);   // x rows m0+0..15
    bf16x16 a1  = ldfrag(abase1 + kb);   // x rows m0+16..31
    bf16x16 bw0 = ldfrag(wbase0 + kb);   // W1 cols n0+0..15   (L2-resident)
    bf16x16 bw1 = ldfrag(wbase1 + kb);   // W1 cols n0+16..31
    acc[0][0] = __builtin_amdgcn_wmma_f32_16x16x32_bf16(false, a0, false, bw0, (short)0, acc[0][0], false, false);
    acc[0][1] = __builtin_amdgcn_wmma_f32_16x16x32_bf16(false, a0, false, bw1, (short)0, acc[0][1], false, false);
    acc[1][0] = __builtin_amdgcn_wmma_f32_16x16x32_bf16(false, a1, false, bw0, (short)0, acc[1][0], false, false);
    acc[1][1] = __builtin_amdgcn_wmma_f32_16x16x32_bf16(false, a1, false, bw1, (short)0, acc[1][1], false, false);
  }

  // fused epilogue: pre += b1 ; h = tanh(pre) ; partial logit = sum_n h*W2[n]
  float bb0 = b1[n0 + lrow],      bb1 = b1[n0 + 16 + lrow];
  float ww0 = w2[n0 + lrow],      ww1 = w2[n0 + 16 + lrow];

  #pragma unroll
  for (int mb = 0; mb < 2; ++mb) {
    #pragma unroll
    for (int r = 0; r < 8; ++r) {
      float t = tanh_fast(acc[mb][0][r] + bb0) * ww0
              + tanh_fast(acc[mb][1][r] + bb1) * ww1;
      // reduce over the 16 lanes of this half (they share the row, differ in col)
      t += __shfl_xor(t, 8, 32);
      t += __shfl_xor(t, 4, 32);
      t += __shfl_xor(t, 2, 32);
      t += __shfl_xor(t, 1, 32);
      if (lrow == 0) lpart[wid][mb * 16 + khalf * 8 + r] = t;
    }
  }
  __syncthreads();

  // deterministic cross-wave sum (fixed order) + b2
  if (tid < 32) {
    float s = b2[0];
    #pragma unroll
    for (int w = 0; w < 8; ++w) s += lpart[w][tid];
    int gm = m0 + tid;
    if (gm < n_nodes) logits[gm] = s;
  }
}

// ---------------- kernel 3: per-graph softmax + weighted pooling ----------------
__global__ __launch_bounds__(256) void attn_pool_kernel(
    const float* __restrict__ x, const int* __restrict__ batch,
    const float* __restrict__ logits, float* __restrict__ out, int n_nodes) {
  const int g   = blockIdx.x;
  const int tid = threadIdx.x;
  __shared__ float red[256];

  // batch is sorted: binary-search [s,e) for this graph (redundant per-thread, broadcast loads)
  int lo = 0, hi = n_nodes;
  while (lo < hi) { int mid = (lo + hi) >> 1; if (batch[mid] < g) lo = mid + 1; else hi = mid; }
  const int s = lo;
  hi = n_nodes;
  while (lo < hi) { int mid = (lo + hi) >> 1; if (batch[mid] < g + 1) lo = mid + 1; else hi = mid; }
  const int e = lo;

  // segment max
  float m = -3.402823466e38f;
  for (int i = s + tid; i < e; i += 256) m = fmaxf(m, logits[i]);
  red[tid] = m; __syncthreads();
  for (int off = 128; off > 0; off >>= 1) {
    if (tid < off) red[tid] = fmaxf(red[tid], red[tid + off]);
    __syncthreads();
  }
  m = red[0]; __syncthreads();

  // segment sum of exp
  float z = 0.f;
  for (int i = s + tid; i < e; i += 256) z += __expf(logits[i] - m);
  red[tid] = z; __syncthreads();
  for (int off = 128; off > 0; off >>= 1) {
    if (tid < off) red[tid] += red[tid + off];
    __syncthreads();
  }
  const float rz = (e > s) ? (1.0f / red[0]) : 0.f;
  __syncthreads();

  // weighted pooling: thread owns one hidden column; coalesced x reads
  float acc = 0.f;
  for (int i = s; i < e; ++i) {
    const float w = __expf(logits[i] - m) * rz;   // broadcast load + VALU, cheap
    acc += x[(size_t)i * H + tid] * w;
  }
  out[(size_t)g * H + tid] = acc;   // empty graphs write 0 (matches segment_sum)
}

// ---------------- launcher ----------------
extern "C" void kernel_launch(void* const* d_in, const int* in_sizes, int n_in,
                              void* d_out, int out_size, void* d_ws, size_t ws_size,
                              hipStream_t stream) {
  const float* x     = (const float*)d_in[0];
  const int*   batch = (const int*)  d_in[1];
  // d_in[2] = num_graphs scalar on device (use out_size instead; no sync copies in capture)
  const float* W1    = (const float*)d_in[3];
  const float* b1    = (const float*)d_in[4];
  const float* W2    = (const float*)d_in[5];
  const float* b2    = (const float*)d_in[6];

  const int n_nodes = in_sizes[0] / H;
  const int G       = out_size / H;

  float* logits = (float*)d_ws;
  size_t loff = ((size_t)n_nodes * sizeof(float) + 255) & ~(size_t)255;
  unsigned short* w1t = (unsigned short*)((char*)d_ws + loff);

  convert_w1_kernel<<<H, H, 0, stream>>>(W1, w1t);
  attn_logits_kernel<<<(n_nodes + 31) / 32, 256, 0, stream>>>(x, w1t, b1, W2, b2, logits, n_nodes);
  attn_pool_kernel<<<G, 256, 0, stream>>>(x, batch, logits, (float*)d_out, n_nodes);
}
